// EncoderLayer_57690000720047
// MI455X (gfx1250) — compile-verified, matrix operand order checked by reading, not run on hardware
//
#include <hip/hip_runtime.h>
#include <cstdint>
#include <cstddef>

// ---------------------------------------------------------------------------
// Types for gfx1250 WMMA (wave32)
// ---------------------------------------------------------------------------
typedef __attribute__((ext_vector_type(16))) __bf16    v16bf;
typedef __attribute__((ext_vector_type(8)))  float     v8f;
typedef __attribute__((ext_vector_type(4)))  uint32_t  v4u;

union Frag {
    v16bf bf;
    v4u   q[2];
};

__device__ __forceinline__ v8f zero8() {
    v8f z = {0.f, 0.f, 0.f, 0.f, 0.f, 0.f, 0.f, 0.f};
    return z;
}

__device__ __forceinline__ uint16_t f2bf(float f) {
    uint32_t u = __float_as_uint(f);
    uint32_t r = u + 0x7FFFu + ((u >> 16) & 1u);   // round-to-nearest-even
    return (uint16_t)(r >> 16);
}

// Load a 16-bit A/B WMMA fragment: two 16-byte chunks at +0 and +16 elements.
// Caller supplies base = &buf[row*pitch + kh*8] (row = lane&15, kh = lane>>4).
__device__ __forceinline__ v16bf load_frag(const uint16_t* p) {
    Frag f;
    f.q[0] = *(const v4u*)(p);
    f.q[1] = *(const v4u*)(p + 16);
    return f.bf;
}

// ---------------------------------------------------------------------------
// Problem constants
// ---------------------------------------------------------------------------
constexpr int NBATCH = 2;
constexpr int SEQ    = 2048;
constexpr int HDIM   = 1024;
constexpr int FFNDIM = 4096;
constexpr int NHEADS = 16;
constexpr int DHEAD  = 64;
constexpr int MROWS  = NBATCH * SEQ;   // 4096

// ---------------------------------------------------------------------------
// fp32 -> bf16 convert
// ---------------------------------------------------------------------------
__global__ __launch_bounds__(256)
void cvt_kernel(const float* __restrict__ in, uint16_t* __restrict__ out, size_t n) {
    size_t i      = (size_t)blockIdx.x * blockDim.x + threadIdx.x;
    size_t stride = (size_t)gridDim.x * blockDim.x;
    for (; i < n; i += stride) out[i] = f2bf(in[i]);
}

// ---------------------------------------------------------------------------
// GEMM: C[M,N] = A[M,K](bf16) * B[K,N](bf16) + bias, optional ReLU
// Block: 256 threads (8 waves), tile 128x128, K-step 32, double-buffered LDS.
// A tile is staged with GLOBAL_LOAD_ASYNC_TO_LDS_B128 (ASYNCcnt path);
// B tile is pipelined global->VGPR before WMMA, VGPR->LDS transpose after.
// Wave (wm 0..3, wn 0..1) computes a 32x64 sub-tile = 2x4 WMMA accumulators.
// ---------------------------------------------------------------------------
#define BM 128
#define BN 128
#define BK 32
#define LDP 48   // LDS pitch in bf16 elems (96B rows -> 16B aligned chunks)

template<bool STORE_BF16, bool RELU>
__global__ __launch_bounds__(256)
void gemm_bf16_kernel(const uint16_t* __restrict__ A, const uint16_t* __restrict__ B,
                      const float* __restrict__ bias, void* __restrict__ C,
                      int M, int N, int K)
{
    __shared__ uint16_t As[2][BM * LDP];   // row-major M x K tile (double buffer)
    __shared__ uint16_t Bs[2][BN * LDP];   // transposed: N-major, K-contiguous

    const int tid  = threadIdx.x;
    const int lane = tid & 31;
    const int wave = tid >> 5;     // 0..7
    const int wm   = wave >> 1;    // 0..3 -> 32-row group
    const int wn   = wave & 1;     // 0..1 -> 64-col group
    const int bm0  = blockIdx.y * BM;
    const int bn0  = blockIdx.x * BN;
    const int fr   = lane & 15;
    const int kh   = lane >> 4;

    // --- staging helpers ----------------------------------------------------
    // A tile: 128x32 bf16 = 512 x 16B chunks; async DMA straight into LDS.
    auto stage_a_async = [&](int k0, int buf) {
#pragma unroll
        for (int t = 0; t < 2; t++) {
            int c   = tid + t * 256;
            int row = c >> 2;      // 4 chunks per row
            int q   = c & 3;
            uint64_t ga = (uint64_t)(uintptr_t)(A + (size_t)(bm0 + row) * K + k0 + q * 8);
            uint32_t la = (uint32_t)(uintptr_t)(&As[buf][row * LDP + q * 8]);
            asm volatile("global_load_async_to_lds_b128 %0, %1, off"
                         :: "v"(la), "v"(ga) : "memory");
        }
    };
    // B tile: rows k (32) x cols n (128); load to VGPRs (issued early) ...
    auto load_b_regs = [&](int k0, v4u* br) {
#pragma unroll
        for (int t = 0; t < 2; t++) {
            int c  = tid + t * 256;
            int kr = c >> 4;       // 16 chunks per k-row
            int nq = c & 15;
            br[t] = *(const v4u*)(B + (size_t)(k0 + kr) * N + bn0 + nq * 8);
        }
    };
    // ... and scatter transposed into LDS (issued after the WMMA block).
    auto store_b_lds = [&](const v4u* br, int buf) {
#pragma unroll
        for (int t = 0; t < 2; t++) {
            int c  = tid + t * 256;
            int kr = c >> 4;
            int nq = c & 15;
            const uint16_t* e = (const uint16_t*)&br[t];
#pragma unroll
            for (int j = 0; j < 8; j++) Bs[buf][(nq * 8 + j) * LDP + kr] = e[j];
        }
    };

    v8f acc[2][4];
#pragma unroll
    for (int i = 0; i < 2; i++)
#pragma unroll
        for (int j = 0; j < 4; j++) acc[i][j] = zero8();

    // --- prologue: fill buffer 0 --------------------------------------------
    stage_a_async(0, 0);
    {
        v4u br[2];
        load_b_regs(0, br);
        store_b_lds(br, 0);
    }
    asm volatile("s_wait_asynccnt 0x0" ::: "memory");
    __syncthreads();

    // --- main pipelined loop ------------------------------------------------
    int p = 0;
    for (int k0 = 0; k0 < K; k0 += BK, p ^= 1) {
        const bool more = (k0 + BK) < K;
        v4u br[2];
        if (more) {
            stage_a_async(k0 + BK, p ^ 1);   // async DMA next A tile
            load_b_regs(k0 + BK, br);        // issue next B global loads
        }
        if (k0 + 2 * BK < K) {               // warm L2 one tile further out
            __builtin_prefetch((const void*)(A + (size_t)(bm0 + (tid >> 1)) * K + k0 + 2 * BK), 0, 1);
            __builtin_prefetch((const void*)(B + (size_t)(k0 + 2 * BK + (tid >> 3)) * N + bn0 + (tid & 7) * 16), 0, 1);
        }

        // --- fragments from current buffer ----------------------------------
        v16bf afr[2], bfr[4];
#pragma unroll
        for (int mi = 0; mi < 2; mi++)
            afr[mi] = load_frag(&As[p][(wm * 32 + mi * 16 + fr) * LDP + kh * 8]);
#pragma unroll
        for (int ni = 0; ni < 4; ni++)
            bfr[ni] = load_frag(&Bs[p][(wn * 64 + ni * 16 + fr) * LDP + kh * 8]);

#pragma unroll
        for (int mi = 0; mi < 2; mi++)
#pragma unroll
            for (int ni = 0; ni < 4; ni++)
                acc[mi][ni] = __builtin_amdgcn_wmma_f32_16x16x32_bf16(
                    false, afr[mi], false, bfr[ni], (short)0, acc[mi][ni], false, false);

        if (more) {
            store_b_lds(br, p ^ 1);          // finish B staging for next tile
            asm volatile("s_wait_asynccnt 0x0" ::: "memory");
        }
        __syncthreads();                     // next-buffer ready / this-buffer free
    }

    // --- epilogue: C layout lane(n=fr), vgpr r -> row 8*kh + r ---------------
#pragma unroll
    for (int mi = 0; mi < 2; mi++) {
#pragma unroll
        for (int ni = 0; ni < 4; ni++) {
            int col = bn0 + wn * 64 + ni * 16 + fr;
            float bv = bias[col];
#pragma unroll
            for (int r = 0; r < 8; r++) {
                int row = bm0 + wm * 32 + mi * 16 + kh * 8 + r;
                float v = acc[mi][ni][r] + bv;
                if (RELU) v = v > 0.f ? v : 0.f;
                if constexpr (STORE_BF16)
                    ((uint16_t*)C)[(size_t)row * N + col] = f2bf(v);
                else
                    ((float*)C)[(size_t)row * N + col] = v;
            }
        }
    }
}

// ---------------------------------------------------------------------------
// Flash-style attention. Grid (SEQ/64, NHEADS, NBATCH), 128 threads (4 waves).
// Wave w handles 16 query rows. Streams 64-key tiles with online softmax.
// Score quirk from the reference: scale = 1/H (not 1/sqrt(d)).
// ---------------------------------------------------------------------------
__global__ __launch_bounds__(128)
void attn_kernel(const uint16_t* __restrict__ Qb, const uint16_t* __restrict__ Kb,
                 const uint16_t* __restrict__ Vb, const int* __restrict__ km,
                 uint16_t* __restrict__ ctx)
{
    const int b = blockIdx.z, h = blockIdx.y, qb = blockIdx.x;
    const int tid = threadIdx.x, lane = tid & 31, wave = tid >> 5;
    const int fr = lane & 15, kh = lane >> 4;
    const int q0 = qb * 64 + wave * 16;
    const size_t base = (size_t)b * SEQ * HDIM + (size_t)h * DHEAD;

    __shared__ uint16_t Vt[DHEAD * 72];     // V^T tile: [d][key], shared by all waves
    __shared__ uint16_t Pl[4][16 * 72];     // per-wave P tile: [qrow][key]

    // Q fragments (A operand, 16x32 bf16, d split in two chunks) - load once
    v16bf qf[2];
#pragma unroll
    for (int kc = 0; kc < 2; kc++)
        qf[kc] = load_frag(Qb + base + (size_t)(q0 + fr) * HDIM + kc * 32 + kh * 8);

    float rmax[8], rsum[8];
#pragma unroll
    for (int r = 0; r < 8; r++) { rmax[r] = -1e30f; rsum[r] = 0.f; }
    v8f cacc[4];
#pragma unroll
    for (int i = 0; i < 4; i++) cacc[i] = zero8();

    const float scale = 1.0f / (float)HDIM;

    for (int kt = 0; kt < SEQ; kt += 64) {
        // --- stage V^T tile (block-wide) ------------------------------------
        for (int c = tid; c < (64 * DHEAD) / 8; c += 128) {
            int key = c >> 3, dq = c & 7;
            v4u v = *(const v4u*)(Vb + base + (size_t)(kt + key) * HDIM + dq * 8);
            const uint16_t* e = (const uint16_t*)&v;
#pragma unroll
            for (int j = 0; j < 8; j++) Vt[(dq * 8 + j) * 72 + key] = e[j];
        }

        // --- scores S = Q K^T : K rows are already N-major/K-contiguous -----
        v8f s[4];
#pragma unroll
        for (int i = 0; i < 4; i++) s[i] = zero8();
#pragma unroll
        for (int ni = 0; ni < 4; ni++) {
#pragma unroll
            for (int kc = 0; kc < 2; kc++) {
                v16bf kf = load_frag(Kb + base + (size_t)(kt + ni * 16 + fr) * HDIM + kc * 32 + kh * 8);
                s[ni] = __builtin_amdgcn_wmma_f32_16x16x32_bf16(
                    false, qf[kc], false, kf, (short)0, s[ni], false, false);
            }
        }
        // scale + key mask (C layout: col = fr -> key index)
#pragma unroll
        for (int ni = 0; ni < 4; ni++) {
            float mv = km[b * SEQ + kt + ni * 16 + fr] ? 0.f : -1e30f;
#pragma unroll
            for (int r = 0; r < 8; r++) s[ni][r] = s[ni][r] * scale + mv;
        }

        // --- online softmax (rows m = 8*kh + r; reduce across 16-lane half) -
        float nm[8], fcorr[8], tsum[8];
#pragma unroll
        for (int r = 0; r < 8; r++) {
            float v = fmaxf(fmaxf(s[0][r], s[1][r]), fmaxf(s[2][r], s[3][r]));
            v = fmaxf(v, __shfl_xor(v, 1, 16));
            v = fmaxf(v, __shfl_xor(v, 2, 16));
            v = fmaxf(v, __shfl_xor(v, 4, 16));
            v = fmaxf(v, __shfl_xor(v, 8, 16));
            nm[r]    = fmaxf(rmax[r], v);
            fcorr[r] = __expf(rmax[r] - nm[r]);
            rmax[r]  = nm[r];
            tsum[r]  = 0.f;
        }
#pragma unroll
        for (int ni = 0; ni < 4; ni++)
#pragma unroll
            for (int r = 0; r < 8; r++) {
                float p = __expf(s[ni][r] - nm[r]);
                s[ni][r] = p;
                tsum[r] += p;
            }
#pragma unroll
        for (int r = 0; r < 8; r++) {
            float v = tsum[r];
            v += __shfl_xor(v, 1, 16);
            v += __shfl_xor(v, 2, 16);
            v += __shfl_xor(v, 4, 16);
            v += __shfl_xor(v, 8, 16);
            rsum[r] = rsum[r] * fcorr[r] + v;
        }
#pragma unroll
        for (int i = 0; i < 4; i++)
#pragma unroll
            for (int r = 0; r < 8; r++) cacc[i][r] *= fcorr[r];

        // --- transpose P into A-fragment layout via LDS ---------------------
        uint16_t* pl = &Pl[wave][0];
#pragma unroll
        for (int ni = 0; ni < 4; ni++)
#pragma unroll
            for (int r = 0; r < 8; r++)
                pl[(kh * 8 + r) * 72 + ni * 16 + fr] = f2bf(s[ni][r]);

        __syncthreads();   // Vt + Pl visible

        // --- ctx += P (16x64) * V (64x64) -----------------------------------
        v16bf pf[2];
#pragma unroll
        for (int kc = 0; kc < 2; kc++)
            pf[kc] = load_frag(pl + fr * 72 + kc * 32 + kh * 8);
#pragma unroll
        for (int nd = 0; nd < 4; nd++) {
#pragma unroll
            for (int kc = 0; kc < 2; kc++) {
                v16bf vf = load_frag(&Vt[(nd * 16 + fr) * 72 + kc * 32 + kh * 8]);
                cacc[nd] = __builtin_amdgcn_wmma_f32_16x16x32_bf16(
                    false, pf[kc], false, vf, (short)0, cacc[nd], false, false);
            }
        }
        __syncthreads();   // done reading before next tile restages LDS
    }

    // --- normalize and write ctx in (N,L,H) concat-head layout (bf16) -------
#pragma unroll
    for (int nd = 0; nd < 4; nd++)
#pragma unroll
        for (int r = 0; r < 8; r++) {
            float v = cacc[nd][r] / rsum[r];
            ctx[base + (size_t)(q0 + kh * 8 + r) * HDIM + nd * 16 + fr] = f2bf(v);
        }
}

// ---------------------------------------------------------------------------
// Residual + LayerNorm over H=1024. One block (256 threads) per row.
// Emits fp32 and optionally a bf16 copy for the next GEMM.
// ---------------------------------------------------------------------------
template<bool EMIT_BF>
__global__ __launch_bounds__(256)
void ln_kernel(const float* __restrict__ xa, const float* __restrict__ xb,
               const float* __restrict__ g, const float* __restrict__ bt,
               float* __restrict__ of, uint16_t* __restrict__ ob)
{
    const int t = threadIdx.x;
    const size_t roff = (size_t)blockIdx.x * HDIM;
    __shared__ float red[8];

    float v[4];
    float s = 0.f;
#pragma unroll
    for (int i = 0; i < 4; i++) {
        int c = t + i * 256;
        float x = xa[roff + c] + xb[roff + c];
        v[i] = x; s += x;
    }
#pragma unroll
    for (int o = 1; o < 32; o <<= 1) s += __shfl_xor(s, o, 32);
    if ((t & 31) == 0) red[t >> 5] = s;
    __syncthreads();
    float tot = 0.f;
#pragma unroll
    for (int i = 0; i < 8; i++) tot += red[i];
    float mu = tot * (1.0f / HDIM);
    __syncthreads();

    float s2 = 0.f;
#pragma unroll
    for (int i = 0; i < 4; i++) { float d = v[i] - mu; s2 += d * d; }
#pragma unroll
    for (int o = 1; o < 32; o <<= 1) s2 += __shfl_xor(s2, o, 32);
    if ((t & 31) == 0) red[t >> 5] = s2;
    __syncthreads();
    float tv = 0.f;
#pragma unroll
    for (int i = 0; i < 8; i++) tv += red[i];
    float inv = rsqrtf(tv * (1.0f / HDIM) + 1e-5f);

#pragma unroll
    for (int i = 0; i < 4; i++) {
        int c = t + i * 256;
        float y = (v[i] - mu) * inv * g[c] + bt[c];
        of[roff + c] = y;
        if constexpr (EMIT_BF) ob[roff + c] = f2bf(y);
    }
}

// ---------------------------------------------------------------------------
// Host orchestration
// ---------------------------------------------------------------------------
extern "C" void kernel_launch(void* const* d_in, const int* in_sizes, int n_in,
                              void* d_out, int out_size, void* d_ws, size_t ws_size,
                              hipStream_t stream)
{
    (void)in_sizes; (void)n_in; (void)out_size; (void)ws_size;
    const float* src = (const float*)d_in[0];
    const int*   km  = (const int*)  d_in[1];
    const float* Wq  = (const float*)d_in[2];
    const float* bq  = (const float*)d_in[3];
    const float* Wk  = (const float*)d_in[4];
    const float* bk  = (const float*)d_in[5];
    const float* Wv  = (const float*)d_in[6];
    const float* bv  = (const float*)d_in[7];
    const float* Wo  = (const float*)d_in[8];
    const float* bo  = (const float*)d_in[9];
    const float* g1  = (const float*)d_in[10];
    const float* bn1 = (const float*)d_in[11];
    const float* W1  = (const float*)d_in[12];
    const float* bf1 = (const float*)d_in[13];
    const float* W2  = (const float*)d_in[14];
    const float* bf2 = (const float*)d_in[15];
    const float* g2  = (const float*)d_in[16];
    const float* bn2 = (const float*)d_in[17];

    char* ws = (char*)d_ws;
    size_t off = 0;
    auto alloc = [&](size_t bytes) {
        char* p = ws + off;
        off = (off + bytes + 255) & ~(size_t)255;
        return p;
    };

    uint16_t* srcbf = (uint16_t*)alloc((size_t)MROWS * HDIM * 2);
    uint16_t* Wqb   = (uint16_t*)alloc((size_t)HDIM * HDIM * 2);
    uint16_t* Wkb   = (uint16_t*)alloc((size_t)HDIM * HDIM * 2);
    uint16_t* Wvb   = (uint16_t*)alloc((size_t)HDIM * HDIM * 2);
    uint16_t* Wob   = (uint16_t*)alloc((size_t)HDIM * HDIM * 2);
    uint16_t* W1b   = (uint16_t*)alloc((size_t)HDIM * FFNDIM * 2);
    uint16_t* W2b   = (uint16_t*)alloc((size_t)FFNDIM * HDIM * 2);
    uint16_t* Qb    = (uint16_t*)alloc((size_t)MROWS * HDIM * 2);
    uint16_t* Kb    = (uint16_t*)alloc((size_t)MROWS * HDIM * 2);
    uint16_t* Vb    = (uint16_t*)alloc((size_t)MROWS * HDIM * 2);
    uint16_t* Cb    = (uint16_t*)alloc((size_t)MROWS * HDIM * 2);
    float*    mhsa  = (float*)   alloc((size_t)MROWS * HDIM * 4);
    float*    x1f   = (float*)   alloc((size_t)MROWS * HDIM * 4);
    uint16_t* x1b   = (uint16_t*)alloc((size_t)MROWS * HDIM * 2);
    uint16_t* h1b   = (uint16_t*)alloc((size_t)MROWS * FFNDIM * 2);
    float*    ffn   = (float*)   alloc((size_t)MROWS * HDIM * 4);

    // --- fp32 -> bf16 conversions -------------------------------------------
    cvt_kernel<<<1024, 256, 0, stream>>>(src, srcbf, (size_t)MROWS * HDIM);
    cvt_kernel<<<512,  256, 0, stream>>>(Wq, Wqb, (size_t)HDIM * HDIM);
    cvt_kernel<<<512,  256, 0, stream>>>(Wk, Wkb, (size_t)HDIM * HDIM);
    cvt_kernel<<<512,  256, 0, stream>>>(Wv, Wvb, (size_t)HDIM * HDIM);
    cvt_kernel<<<512,  256, 0, stream>>>(Wo, Wob, (size_t)HDIM * HDIM);
    cvt_kernel<<<1024, 256, 0, stream>>>(W1, W1b, (size_t)HDIM * FFNDIM);
    cvt_kernel<<<1024, 256, 0, stream>>>(W2, W2b, (size_t)FFNDIM * HDIM);

    dim3 blk(256);
    dim3 gH(HDIM / BN, MROWS / BM);       // (8, 32)
    dim3 gF(FFNDIM / BN, MROWS / BM);     // (32, 32)

    // --- Q/K/V projections (bf16 out for attention) -------------------------
    gemm_bf16_kernel<true,  false><<<gH, blk, 0, stream>>>(srcbf, Wqb, bq, Qb, MROWS, HDIM, HDIM);
    gemm_bf16_kernel<true,  false><<<gH, blk, 0, stream>>>(srcbf, Wkb, bk, Kb, MROWS, HDIM, HDIM);
    gemm_bf16_kernel<true,  false><<<gH, blk, 0, stream>>>(srcbf, Wvb, bv, Vb, MROWS, HDIM, HDIM);

    // --- attention ----------------------------------------------------------
    attn_kernel<<<dim3(SEQ / 64, NHEADS, NBATCH), dim3(128), 0, stream>>>(Qb, Kb, Vb, km, Cb);

    // --- output projection + LN1 -------------------------------------------
    gemm_bf16_kernel<false, false><<<gH, blk, 0, stream>>>(Cb, Wob, bo, mhsa, MROWS, HDIM, HDIM);
    ln_kernel<true><<<MROWS, 256, 0, stream>>>(mhsa, src, g1, bn1, x1f, x1b);

    // --- FFN + LN2 ----------------------------------------------------------
    gemm_bf16_kernel<true,  true ><<<gF, blk, 0, stream>>>(x1b, W1b, bf1, h1b, MROWS, FFNDIM, HDIM);
    gemm_bf16_kernel<false, false><<<gH, blk, 0, stream>>>(h1b, W2b, bf2, ffn, MROWS, HDIM, FFNDIM);
    ln_kernel<false><<<MROWS, 256, 0, stream>>>(ffn, x1f, g2, bn2, (float*)d_out, nullptr);
}
